// SNRLoss_36309653520997
// MI455X (gfx1250) — matrix operand order repaired and checked by reading, not compiled, not run on hardware
//
#include <hip/hip_runtime.h>
#include <hip/hip_bf16.h>
#include <math.h>

typedef __attribute__((ext_vector_type(2)))  _Float16 h2;
typedef __attribute__((ext_vector_type(8)))  _Float16 h8;
typedef __attribute__((ext_vector_type(16))) _Float16 h16;
typedef __attribute__((ext_vector_type(4)))  float v4f;
typedef __attribute__((ext_vector_type(8)))  float v8f;

#define NROW   1024
#define NSAMP  32768
#define PDIM   256          // n = p + 256*q
#define QDIM   128
#define ADIM   128          // k = a + 128*c
#define CLO    8            // c in [8,56) covers k in [1024,7168)
#define CCNT   48
#define KMIN   1092
#define KMAX   6827
#define DENOMF 5732.0f
#define FSTEP  6.103515625e-4f   // (Fs/2)/16384 = 10/16384, exact in f32

// ---- LDS layout (byte offsets; all b128 bases 16B aligned) ----
// Wc16/Ws16[a][q] f16, row stride 272B (136 halves): stage-1 B fragments
#define WC16_B   0              // 128*272 = 34816
#define WS16_B   34816          // 34816
// Hc16/Hs16[cc][p] f16, row stride 528B (264 halves): stage-2 A fragments
#define HC16_B   69632          // 48*528 = 25344
#define HS16_B   94976          // 25344
// A1h[ph][q] f16 x-tile, row stride 272B
#define A1H_B    120320         // 16*272 = 4352
// Zr/Zi[a][p_local 0..31] f16, row stride 80B: stage-2 B fragments (dbl-buffer)
#define ZR16_B   124672         // 128*80 = 10240
#define ZI16_B   134912         // 10240
#define LDS_BYTES 145152        // ~141.8 KB -> 2 workgroups per WGP

// Overlay after the p-loop (W tables are dead by then)
#define XR_B     0              // 48 x 132 f32
#define XI_B     25344          // 48 x 132 f32
#define X_STR    132
#define RED_B    50688          // 256 f32

static __device__ __forceinline__ v8f wmma32(h16 a, h16 b, v8f c) {
    return __builtin_amdgcn_wmma_f32_16x16x32_f16(false, a, false, b,
                                                  (short)0, c, false, false);
}
static __device__ __forceinline__ h16 ld_frag(const char* lo, const char* hi) {
    h8 l = *(const h8*)lo;
    h8 h = *(const h8*)hi;
    return __builtin_shufflevector(l, h, 0,1,2,3,4,5,6,7,8,9,10,11,12,13,14,15);
}

extern "C" __global__ void __launch_bounds__(256)
snr_rows_kernel(const float* __restrict__ x, const float* __restrict__ targets,
                float* __restrict__ row_loss)
{
    extern __shared__ char ldsc[];
    const int tid  = threadIdx.x;
    const int b    = blockIdx.x;
    const int lane = tid & 31;
    const int w    = tid >> 5;        // wave id 0..7, owns a-tile [16w,16w+16)
    const int kh   = lane >> 4;       // 0/1 : K half for WMMA frags
    const int l15  = lane & 15;
    const int aC   = w * 16 + l15;    // this lane's 'a' column (B frag / C-D col)

    // ---- Phase A: f16 trig tables in WMMA-ready layouts ----
    {
        const float s128 = 6.2831853071795864769f / 128.0f;
        for (int idx = tid; idx < 128 * 128; idx += 256) {
            const int a = idx >> 7, q = idx & 127;
            float sv, cv;
            sincosf((float)((q * a) & 127) * s128, &sv, &cv);
            *(_Float16*)(ldsc + WC16_B + a * 272 + q * 2) = (_Float16)cv;
            *(_Float16*)(ldsc + WS16_B + a * 272 + q * 2) = (_Float16)sv;
        }
        const float s256 = 6.2831853071795864769f / 256.0f;
        for (int idx = tid; idx < CCNT * 256; idx += 256) {
            const int cc = idx >> 8, p = idx & 255;
            float sv, cv;
            sincosf((float)(((cc + CLO) * p) & 255) * s256, &sv, &cv);
            *(_Float16*)(ldsc + HC16_B + cc * 528 + p * 2) = (_Float16)cv;
            *(_Float16*)(ldsc + HS16_B + cc * 528 + p * 2) = (_Float16)sv;
        }
    }

    const v8f vzero = {0.f,0.f,0.f,0.f,0.f,0.f,0.f,0.f};
    v8f accR[3], accI[3];
    for (int ct = 0; ct < 3; ++ct) { accR[ct] = vzero; accI[ct] = vzero; }

    // per-lane incremental twiddle rotator for theta = 2*pi*aC*p/32768
    // (aC*p <= 127*255 < 32768, so no wrap is ever needed)
    const float tw = 6.2831853071795864769f / 32768.0f;
    float c1, s1, c9, s9, ctw, stw;
    {
        const float del = (float)aC * tw;
        sincosf(del, &s1, &c1);                 // step p -> p+1
        sincosf(9.0f * del, &s9, &c9);          // step p -> p+9 (tile hop)
        sincosf((float)(aC * 8 * kh) * tw, &stw, &ctw);   // phase at first p
    }

    const float* __restrict__ xrow = x + (size_t)b * NSAMP;
    __syncthreads();

    for (int pt = 0; pt < 16; ++pt) {
        const int p0 = pt * 16;
        // ---- B1: stage x tile (f32 b128 loads -> f16 LDS), prefetch next ----
        for (int idx = tid; idx < 512; idx += 256) {
            const int q  = idx >> 2;
            const int pi = (idx & 3) * 4;
            const v4f v = *(const v4f*)(xrow + (p0 + pi) + PDIM * q);
            *(_Float16*)(ldsc + A1H_B + (pi + 0) * 272 + q * 2) = (_Float16)v.x;
            *(_Float16*)(ldsc + A1H_B + (pi + 1) * 272 + q * 2) = (_Float16)v.y;
            *(_Float16*)(ldsc + A1H_B + (pi + 2) * 272 + q * 2) = (_Float16)v.z;
            *(_Float16*)(ldsc + A1H_B + (pi + 3) * 272 + q * 2) = (_Float16)v.w;
            if (pt < 15)
                __builtin_prefetch(xrow + (p0 + 16 + pi) + PDIM * q, 0, 0);
        }
        __syncthreads();

        // ---- B2: stage-1 128-pt DFT over q (K=128 = 4 WMMA k-chunks) ----
        v8f gr = vzero, gi = vzero;   // gi = +sum x*sin (true imag = -gi)
        for (int s = 0; s < 4; ++s) {
            const char* ab = ldsc + A1H_B + l15 * 272 + s * 64 + 32 * kh; // A layout
            const h16 av = ld_frag(ab, ab + 16);
            const char* cb = ldsc + WC16_B + aC * 272 + s * 64 + 32 * kh; // B layout
            const h16 bc = ld_frag(cb, cb + 16);
            const char* sb = ldsc + WS16_B + aC * 272 + s * 64 + 32 * kh;
            const h16 bs = ld_frag(sb, sb + 16);
            gr = wmma32(av, bc, gr);
            gi = wmma32(av, bs, gi);
        }
        // twiddle: Z = (gr - i*gi) * (cos - i*sin), rotator advances phase
        for (int r = 0; r < 8; r += 2) {
            const float zr0 =   gr[r] * ctw - gi[r] * stw;
            const float zi0 = -(gr[r] * stw + gi[r] * ctw);
            { const float cn = ctw * c1 - stw * s1; stw = stw * c1 + ctw * s1; ctw = cn; }
            const float zr1 =   gr[r + 1] * ctw - gi[r + 1] * stw;
            const float zi1 = -(gr[r + 1] * stw + gi[r + 1] * ctw);
            if (r < 6) { const float cn = ctw * c1 - stw * s1; stw = stw * c1 + ctw * s1; ctw = cn; }
            else       { const float cn = ctw * c9 - stw * s9; stw = stw * c9 + ctw * s9; ctw = cn; }
            const int ploc = (pt & 1) * 16 + 8 * kh + r;    // even
            h2 hzr, hzi;
            hzr.x = (_Float16)zr0; hzr.y = (_Float16)zr1;
            hzi.x = (_Float16)zi0; hzi.y = (_Float16)zi1;
            *(h2*)(ldsc + ZR16_B + aC * 80 + ploc * 2) = hzr;
            *(h2*)(ldsc + ZI16_B + aC * 80 + ploc * 2) = hzi;
        }
        __syncthreads();

        // ---- B3: stage-2 partial 256-pt DFT, K=32 (two p-tiles) per shot ----
        if (pt & 1) {
            const int d = pt >> 1;                 // K chunk = p in [32d, 32d+32)
            const char* zr = ldsc + ZR16_B + aC * 80 + 32 * kh;   // B layout
            const h16 bzr = ld_frag(zr, zr + 16);
            const char* zi = ldsc + ZI16_B + aC * 80 + 32 * kh;
            const h16 bzi = ld_frag(zi, zi + 16);
            for (int ct = 0; ct < 3; ++ct) {
                const int row = (16 * ct + l15) * 528 + 64 * d + 32 * kh;  // A layout
                const char* hb = ldsc + HC16_B + row;
                const h16 hc = ld_frag(hb, hb + 16);
                const char* sb = ldsc + HS16_B + row;
                const h16 hs = ld_frag(sb, sb + 16);
                const h16 hsn = -hs;
                // Xr += cos*Zr + sin*Zi ;  Xi += cos*Zi - sin*Zr
                accR[ct] = wmma32(hc,  bzr, accR[ct]);
                accR[ct] = wmma32(hs,  bzi, accR[ct]);
                accI[ct] = wmma32(hc,  bzi, accI[ct]);
                accI[ct] = wmma32(hsn, bzr, accI[ct]);
            }
            __syncthreads();
        }
    }

    // ---- Phase C: dump accumulators into overlay (k = a + 128*(CLO+cc)) ----
    float* __restrict__ xrF = (float*)(ldsc + XR_B);
    float* __restrict__ xiF = (float*)(ldsc + XI_B);
    for (int ct = 0; ct < 3; ++ct) {
        for (int r = 0; r < 8; ++r) {
            const int cc = 16 * ct + r + 8 * kh;
            xrF[cc * X_STR + aC] = accR[ct][r];
            xiF[cc * X_STR + aC] = accI[ct][r];
        }
    }
    __syncthreads();

    // ---- Phase D: band power sum (k = 1024 + idx), deterministic tree ----
    float* __restrict__ red = (float*)(ldsc + RED_B);
    float part = 0.f;
    for (int idx = tid; idx < CCNT * ADIM; idx += 256) {
        const int cc = idx >> 7;
        const int a  = idx & 127;
        const int k  = 1024 + idx;
        if (k >= KMIN && k < KMAX) {
            const float xr = xrF[cc * X_STR + a];
            const float xi = xiF[cc * X_STR + a];
            part += xr * xr + xi * xi;
        }
    }
    red[tid] = part;
    __syncthreads();
    for (int off = 128; off > 0; off >>= 1) {
        if (tid < off) red[tid] += red[tid + off];
        __syncthreads();
    }
    if (tid == 0) {
        const float S = red[0];
        const float t = targets[b];
        // ref_idx = argmin_j |j*FSTEP - t| (first minimum, like jnp.argmin)
        int j0 = (int)(t * 1638.4f);
        int best = (j0 > 0) ? (j0 - 1) : 0;
        float bd = fabsf((float)best * FSTEP - t);
        for (int j = best + 1; j <= j0 + 2 && j <= 16384; ++j) {
            const float d = fabsf((float)j * FSTEP - t);
            if (d < bd) { bd = d; best = j; }
        }
        const int r = best;   // in [1311,6390] -> r-1..r+1 inside [1024,7168)
        float P3[3];
        for (int o = 0; o < 3; ++o) {
            const int k  = r - 1 + o;
            const int cc = (k >> 7) - CLO;
            const int a  = k & 127;
            const float xr = xrF[cc * X_STR + a];
            const float xi = xiF[cc * X_STR + a];
            P3[o] = xr * xr + xi * xi;
        }
        const float other = (S - P3[0] - P3[1] - P3[2]) / DENOMF;
        row_loss[b] = -10.0f * log10f(P3[1] / other);
    }
}

extern "C" __global__ void __launch_bounds__(256)
snr_mean_kernel(const float* __restrict__ row_loss, float* __restrict__ out)
{
    __shared__ float red[256];
    float s = 0.f;
    for (int i = threadIdx.x; i < NROW; i += 256) s += row_loss[i];
    red[threadIdx.x] = s;
    __syncthreads();
    for (int off = 128; off > 0; off >>= 1) {
        if (threadIdx.x < off) red[threadIdx.x] += red[threadIdx.x + off];
        __syncthreads();
    }
    if (threadIdx.x == 0) out[0] = red[0] * (1.0f / (float)NROW);
}

extern "C" void kernel_launch(void* const* d_in, const int* in_sizes, int n_in,
                              void* d_out, int out_size, void* d_ws, size_t ws_size,
                              hipStream_t stream)
{
    (void)in_sizes; (void)n_in; (void)out_size; (void)ws_size;
    const float* x  = (const float*)d_in[0];   // (1024, 32768) f32
    const float* tg = (const float*)d_in[1];   // (1024, 1) f32
    float* loss = (float*)d_ws;                // 1024 f32 scratch
    snr_rows_kernel<<<NROW, 256, LDS_BYTES, stream>>>(x, tg, loss);
    snr_mean_kernel<<<1, 256, 0, stream>>>(loss, (float*)d_out);
}
